// FavorAttention_52329881534744
// MI455X (gfx1250) — matrix operand order repaired and checked by reading, not compiled, not run on hardware
//
#include <hip/hip_runtime.h>

typedef __attribute__((ext_vector_type(16))) __bf16 v16bf;
typedef __attribute__((ext_vector_type(8)))  __bf16 v8bf;
typedef __attribute__((ext_vector_type(8)))  float  v8f;

#define B_ 4
#define H_ 8
#define D_ 64
#define N_ 8192
#define M_ 256
#define BH_ 32
#define DN_ 0.35355339059327379f   /* 64^-0.25 */
#define RATIO_ 0.0625f             /* 256^-0.5 */
#define EPS_ 1e-6f
#define KV_STRIDE 80               /* 64 dims + col 64 = key_norm + pad */
#define KV_TOTAL (BH_ * M_ * KV_STRIDE)

// ---------- order-preserving float<->uint for atomic max ----------
__device__ __forceinline__ unsigned f32_ord(float f) {
  unsigned b = __float_as_uint(f);
  return (b & 0x80000000u) ? ~b : (b | 0x80000000u);
}
__device__ __forceinline__ float ord_f32(unsigned m) {
  unsigned b = (m & 0x80000000u) ? (m & 0x7fffffffu) : ~m;
  return __uint_as_float(b);
}

__device__ __forceinline__ v8f wmma_bf16(v16bf a, v16bf b, v8f c) {
  return __builtin_amdgcn_wmma_f32_16x16x32_bf16(false, a, false, b,
                                                 (short)0, c, false, false);
}
__device__ __forceinline__ v8f zero8() {
  v8f z = {0.f, 0.f, 0.f, 0.f, 0.f, 0.f, 0.f, 0.f};
  return z;
}

// A fragment of projection, pre-scaled by data_normalizer.
// contiguous k order (B matches): lane row m = mb + l16; K runs [k0+hi*16,+16)
__device__ __forceinline__ v16bf make_a_proj(const float* proj, int mb, int k0,
                                             int l16, int hi) {
  const float* row = proj + (mb + l16) * D_ + k0 + hi * 16;
  float tb[16];
  *(float4*)(tb + 0)  = *(const float4*)(row + 0);
  *(float4*)(tb + 4)  = *(const float4*)(row + 4);
  *(float4*)(tb + 8)  = *(const float4*)(row + 8);
  *(float4*)(tb + 12) = *(const float4*)(row + 12);
  v16bf a;
#pragma unroll
  for (int e = 0; e < 16; e++) a[e] = (__bf16)(tb[e] * DN_);
  return a;
}

// 16 contiguous bf16 (16B-aligned) -> fragment
__device__ __forceinline__ v16bf load_b16(const __bf16* p) {
  union { v16bf v; v8bf h[2]; } u;
  u.h[0] = *(const v8bf*)(p);
  u.h[1] = *(const v8bf*)(p + 8);
  return u.v;
}
// two 8-element contiguous bf16 runs -> A fragment
__device__ __forceinline__ v16bf load_a_split(const __bf16* p0, const __bf16* p1) {
  union { v16bf v; v8bf h[2]; } u;
  u.h[0] = *(const v8bf*)(p0);
  u.h[1] = *(const v8bf*)(p1);
  return u.v;
}

// ---------------- kernel 0: init workspace ----------------
__global__ __launch_bounds__(256) void k0_init(unsigned* kstab, float* kv) {
  int i = blockIdx.x * 256 + threadIdx.x;
  if (i < KV_TOTAL) kv[i] = 0.f;
  if (i < BH_) kstab[i] = f32_ord(-3.0e38f);
}

// ---------------- kernel 1: global max of u_k per (b,h) ----------------
__global__ __launch_bounds__(256) void k1_kmax(const float* __restrict__ key,
                                               const float* __restrict__ proj,
                                               unsigned* __restrict__ kstab) {
  __shared__ __bf16 xT[64 * 72];   // [token][d] bf16
  __shared__ float red[256];
  const int bh = blockIdx.y, tile = blockIdx.x;
  const int t = threadIdx.x, w = t >> 5, lane = t & 31;
  const int l16 = lane & 15, hi = lane >> 4;
  const float* kb = key + (size_t)bh * D_ * N_ + tile * 64;

  v16bf pa[2][2];
#pragma unroll
  for (int mi = 0; mi < 2; mi++)
#pragma unroll
    for (int kk = 0; kk < 2; kk++)
      pa[mi][kk] = make_a_proj(proj, w * 32 + mi * 16, kk * 32, l16, hi);

  { // coalesced load, transpose into LDS as bf16
    int n = t & 63, grp = t >> 6;
#pragma unroll
    for (int i = 0; i < 16; i++) {
      int d = grp * 16 + i;
      xT[n * 72 + d] = (__bf16)kb[(size_t)d * N_ + n];
    }
  }
  __syncthreads();

  float mx = -3.0e38f;
#pragma unroll
  for (int mi = 0; mi < 2; mi++) {
#pragma unroll
    for (int jb = 0; jb < 4; jb++) {
      v8f c = zero8();
#pragma unroll
      for (int kk = 0; kk < 2; kk++)
        c = wmma_bf16(pa[mi][kk],
                      load_b16(&xT[(jb * 16 + l16) * 72 + kk * 32 + hi * 16]), c);
#pragma unroll
      for (int r = 0; r < 8; r++) mx = fmaxf(mx, c[r]);
    }
  }
  red[t] = mx;
  __syncthreads();
  if (t < 64)
    red[t] = fmaxf(fmaxf(red[t], red[t + 64]), fmaxf(red[t + 128], red[t + 192]));
  __syncthreads();
  if (t == 0) {
    float m = red[0];
    for (int i = 1; i < 64; i++) m = fmaxf(m, red[i]);
    atomicMax(&kstab[bh], f32_ord(m));
  }
}

// ---------------- kernel 2: phi_k and kv' = phi_k @ [V^T | 1] ----------------
__global__ __launch_bounds__(256) void k2_kv(const float* __restrict__ key,
                                             const float* __restrict__ value,
                                             const float* __restrict__ proj,
                                             const unsigned* __restrict__ kstab,
                                             float* __restrict__ kv) {
  __shared__ __bf16 xT[64 * 72];    // key tile  [token][d]
  __shared__ __bf16 vT[64 * 72];    // value tile [d][token]  (natural order)
  __shared__ __bf16 phi[256 * 72];  // [m][token]
  __shared__ float sq[256];
  __shared__ float diag[64];
  const int bh = blockIdx.y, split = blockIdx.x;
  const int t = threadIdx.x, w = t >> 5, lane = t & 31;
  const int l16 = lane & 15, hi = lane >> 4;
  const float* keyb = key + (size_t)bh * D_ * N_;
  const float* valb = value + (size_t)bh * D_ * N_;
  float* kvb = kv + (size_t)bh * M_ * KV_STRIDE;
  const float stab = ord_f32(kstab[bh]);

  v16bf pa[2][2];
#pragma unroll
  for (int mi = 0; mi < 2; mi++)
#pragma unroll
    for (int kk = 0; kk < 2; kk++)
      pa[mi][kk] = make_a_proj(proj, w * 32 + mi * 16, kk * 32, l16, hi);

  v16bf bones;  // appended ones-column block (col 64 -> key_norm)
#pragma unroll
  for (int e = 0; e < 16; e++) bones[e] = (l16 == 0) ? (__bf16)1.0f : (__bf16)0.0f;

  v8f acc[2][5];
#pragma unroll
  for (int mi = 0; mi < 2; mi++)
#pragma unroll
    for (int cb = 0; cb < 5; cb++) acc[mi][cb] = zero8();

  for (int it = 0; it < 16; ++it) {
    const int n0 = (split * 16 + it) * 64;
    __syncthreads();
    {
      int n = t & 63, grp = t >> 6;
      float s = 0.f;
#pragma unroll
      for (int i = 0; i < 16; i++) {
        int d = grp * 16 + i;
        float kval = keyb[(size_t)d * N_ + n0 + n];
        s += kval * kval;
        xT[n * 72 + d] = (__bf16)kval;                 // transpose
        vT[d * 72 + n] = (__bf16)valb[(size_t)d * N_ + n0 + n];  // natural
      }
      sq[grp * 64 + n] = s;
    }
    // prefetch next tile (one 128B line per thread per tensor covers 16KB tile)
    if (it + 1 < 16) {
      const size_t pn0 = (size_t)(split * 16 + it + 1) * 64;
      const size_t poff = (size_t)(t >> 1) * N_ + pn0 + (size_t)(t & 1) * 32;
      __builtin_prefetch(keyb + poff, 0, 3);
      __builtin_prefetch(valb + poff, 0, 3);
    }
    __syncthreads();
    if (t < 64) diag[t] = (sq[t] + sq[64 + t] + sq[128 + t] + sq[192 + t]) * 0.0625f;
    __syncthreads();

    // u = (DN*P) @ K_tile ; phi -> LDS (wave-local rows, no barrier needed)
#pragma unroll
    for (int mi = 0; mi < 2; mi++) {
      const int mb = w * 32 + mi * 16;
#pragma unroll
      for (int jb = 0; jb < 4; jb++) {
        v8f c = zero8();
#pragma unroll
        for (int kk = 0; kk < 2; kk++)
          c = wmma_bf16(pa[mi][kk],
                        load_b16(&xT[(jb * 16 + l16) * 72 + kk * 32 + hi * 16]), c);
        const int n = jb * 16 + l16;
        const float dg = diag[n];
#pragma unroll
        for (int r = 0; r < 8; r++) {
          float ph = RATIO_ * (__expf(c[r] - dg - stab) + EPS_);
          phi[(mb + r + hi * 8) * 72 + n] = (__bf16)ph;
        }
      }
    }

    // kv' += phi @ [V^T | 1]
#pragma unroll
    for (int kk2 = 0; kk2 < 2; kk2++) {
      v16bf bf[5];
#pragma unroll
      for (int cb = 0; cb < 4; cb++)
        bf[cb] = load_b16(&vT[(cb * 16 + l16) * 72 + kk2 * 32 + hi * 16]);
      bf[4] = bones;
#pragma unroll
      for (int mi = 0; mi < 2; mi++) {
        const __bf16* ap = &phi[(w * 32 + mi * 16 + l16) * 72 + kk2 * 32 + hi * 16];
        v16bf af = load_a_split(ap, ap + 8);
#pragma unroll
        for (int cb = 0; cb < 5; cb++)
          acc[mi][cb] = wmma_bf16(af, bf[cb], acc[mi][cb]);
      }
    }
  }

#pragma unroll
  for (int mi = 0; mi < 2; mi++)
#pragma unroll
    for (int cb = 0; cb < 5; cb++)
#pragma unroll
      for (int r = 0; r < 8; r++) {
        int m = w * 32 + mi * 16 + r + hi * 8;
        int c = cb * 16 + l16;
        atomicAdd(&kvb[m * KV_STRIDE + c], acc[mi][cb][r]);
      }
}

// ---------------- kernel 3: phi_q, out = phi_q^T @ kv', normalize ----------------
__global__ __launch_bounds__(256) void k3_out(const float* __restrict__ query,
                                              const float* __restrict__ proj,
                                              const float* __restrict__ kv,
                                              float* __restrict__ out) {
  extern __shared__ char smem3[];
  __bf16* xT   = (__bf16*)(smem3);               // 64*72*2   = 9216
  __bf16* phi  = (__bf16*)(smem3 + 9216);        // 64*264*2  = 33792  [token][m]
  __bf16* kvT  = (__bf16*)(smem3 + 43008);       // 80*264*2  = 42240  [col][m]
  float* sq    = (float*)(smem3 + 85248);        // 1024
  float* diag  = (float*)(smem3 + 86272);        // 256
  unsigned* stabu = (unsigned*)(smem3 + 86528);  // 256
  float* norm  = (float*)(smem3 + 86784);        // 256  (total 87040)

  const int bh = blockIdx.y, tile = blockIdx.x;
  const int n0 = tile * 64;
  const int t = threadIdx.x, w = t >> 5, lane = t & 31;
  const int l16 = lane & 15, hi = lane >> 4;
  const float* qb = query + (size_t)bh * D_ * N_;
  const float* kvb = kv + (size_t)bh * M_ * KV_STRIDE;

  if (t < 64) stabu[t] = f32_ord(-3.0e38f);
  for (int i = t; i < M_ * KV_STRIDE; i += 256) {  // stage kv' transposed as bf16
    int m = i / KV_STRIDE, c = i % KV_STRIDE;
    kvT[c * 264 + m] = (__bf16)kvb[i];
  }
  {
    int n = t & 63, grp = t >> 6;
    float s = 0.f;
#pragma unroll
    for (int i = 0; i < 16; i++) {
      int d = grp * 16 + i;
      float v = qb[(size_t)d * N_ + n0 + n];
      s += v * v;
      xT[n * 72 + d] = (__bf16)v;
    }
    sq[grp * 64 + n] = s;
  }
  __syncthreads();
  if (t < 64) diag[t] = (sq[t] + sq[64 + t] + sq[128 + t] + sq[192 + t]) * 0.0625f;

  v16bf pa[2][2];
#pragma unroll
  for (int mi = 0; mi < 2; mi++)
#pragma unroll
    for (int kk = 0; kk < 2; kk++)
      pa[mi][kk] = make_a_proj(proj, w * 32 + mi * 16, kk * 32, l16, hi);

  v8f uacc[2][4];
#pragma unroll
  for (int mi = 0; mi < 2; mi++) {
#pragma unroll
    for (int jb = 0; jb < 4; jb++) {
      v8f c = zero8();
#pragma unroll
      for (int kk = 0; kk < 2; kk++)
        c = wmma_bf16(pa[mi][kk],
                      load_b16(&xT[(jb * 16 + l16) * 72 + kk * 32 + hi * 16]), c);
      uacc[mi][jb] = c;
    }
  }
  // per-token max over all 256 features via LDS atomic max
#pragma unroll
  for (int jb = 0; jb < 4; jb++) {
    float lm = -3.0e38f;
#pragma unroll
    for (int mi = 0; mi < 2; mi++)
#pragma unroll
      for (int r = 0; r < 8; r++) lm = fmaxf(lm, uacc[mi][jb][r]);
    atomicMax(&stabu[jb * 16 + l16], f32_ord(lm));
  }
  __syncthreads();

#pragma unroll
  for (int mi = 0; mi < 2; mi++) {
#pragma unroll
    for (int jb = 0; jb < 4; jb++) {
      const int n = jb * 16 + l16;
      const float dg = diag[n];
      const float st = ord_f32(stabu[n]);
#pragma unroll
      for (int r = 0; r < 8; r++) {
        float ph = RATIO_ * (__expf(uacc[mi][jb][r] - dg - st) + EPS_);
        phi[n * 264 + (w * 32 + mi * 16 + r + hi * 8)] = (__bf16)ph;
      }
    }
  }
  __syncthreads();

  // out' = phi_q^T @ kv'  (20 tiles of 16x16 over 8 waves; K = 256)
  v8f oacc[3];
  oacc[0] = zero8(); oacc[1] = zero8(); oacc[2] = zero8();
#pragma unroll
  for (int ci = 0; ci < 3; ci++) {
    int tt = w + 8 * ci;
    if (tt >= 20) break;  // wave-uniform
    int tb = tt & 3, cb = tt >> 2;
#pragma unroll
    for (int kk = 0; kk < 8; kk++) {
      int k0 = kk * 32;
      const __bf16* ap = &phi[(tb * 16 + l16) * 264 + k0 + hi * 16];
      v16bf af = load_a_split(ap, ap + 8);
      v16bf bf = load_b16(&kvT[(cb * 16 + l16) * 264 + k0 + hi * 16]);
      oacc[ci] = wmma_bf16(af, bf, oacc[ci]);
    }
  }
  // column 64 (cb==4, l16==0) is out_norm; broadcast via LDS
  if (w < 4 && l16 == 0) {
#pragma unroll
    for (int r = 0; r < 8; r++) norm[w * 16 + r + hi * 8] = oacc[2][r];
  }
  __syncthreads();

#pragma unroll
  for (int ci = 0; ci < 3; ci++) {
    int tt = w + 8 * ci;
    if (tt >= 20) break;
    int tb = tt & 3, cb = tt >> 2;
    if (cb < 4) {
      int d = cb * 16 + l16;
#pragma unroll
      for (int r = 0; r < 8; r++) {
        int n = tb * 16 + r + hi * 8;
        out[((size_t)bh * D_ + d) * N_ + n0 + n] = oacc[ci][r] / norm[n];
      }
    }
  }
}

extern "C" void kernel_launch(void* const* d_in, const int* in_sizes, int n_in,
                              void* d_out, int out_size, void* d_ws, size_t ws_size,
                              hipStream_t stream) {
  const float* query = (const float*)d_in[0];
  const float* key   = (const float*)d_in[1];
  const float* value = (const float*)d_in[2];
  const float* proj  = (const float*)d_in[3];
  unsigned* kstab = (unsigned*)d_ws;
  float* kv = (float*)d_ws + 64;  // kv' accumulator [BH][256][80] after 256B header
  float* out = (float*)d_out;

  k0_init<<<(KV_TOTAL + 255) / 256, 256, 0, stream>>>(kstab, kv);
  k1_kmax<<<dim3(128, 32), 256, 0, stream>>>(key, proj, kstab);
  k2_kv<<<dim3(8, 32), 256, 0, stream>>>(key, value, proj, kstab, kv);
  k3_out<<<dim3(128, 32), 256, 87040, stream>>>(query, proj, kv, out);

  (void)in_sizes; (void)n_in; (void)out_size; (void)ws_size;
}
